// SSHA_6528350290308
// MI455X (gfx1250) — matrix operand order repaired and checked by reading, not compiled, not run on hardware
//
#include <hip/hip_runtime.h>
#include <hip/hip_bf16.h>

// ---------------- problem constants ----------------
#define NB   16
#define TT   288
#define VV   25
#define CC   128
#define SS   8
#define ICC  16
#define TVV  (TT * VV)      // 7200
#define UU   75             // W*V
#define UPAD 96             // padded u for clean 32-wide WMMA K-steps
#define EPS  1e-5f

typedef _Float16 v16h __attribute__((ext_vector_type(16)));
typedef _Float16 v8h  __attribute__((ext_vector_type(8)));
typedef float    v8f  __attribute__((ext_vector_type(8)));

__device__ __forceinline__ v8f wmma16(v16h a, v16h b, v8f c) {
  return __builtin_amdgcn_wmma_f32_16x16x32_f16(false, a, false, b, (short)0, c,
                                                false, false);
}
__device__ __forceinline__ v16h mk16(v8h lo, v8h hi) {
  v16h r;
#pragma unroll
  for (int i = 0; i < 8; ++i) { r[i] = lo[i]; r[i + 8] = hi[i]; }
  return r;
}

// ---------------- K0: f32 -> f16 convert (also used for weight packing) -----
__global__ void f32_to_f16_kernel(const float* __restrict__ src,
                                  _Float16* __restrict__ dst, int count4) {
  int i = blockIdx.x * blockDim.x + threadIdx.x;
  if (i < count4) {
    float4 v = reinterpret_cast<const float4*>(src)[i];
    _Float16* d = dst + (size_t)i * 4;
    d[0] = (_Float16)v.x; d[1] = (_Float16)v.y;
    d[2] = (_Float16)v.z; d[3] = (_Float16)v.w;
  }
}

// ---------------- K1/K5: row GEMM  out[n,tv,o] = A[n,tv,:] . W[o,:] ----------
// EP=false: out = conv + bias (f16)
// EP=true : out = leaky( x + BN(conv + bias) )   (f16)
template <bool EP>
__global__ void rowgemm16_kernel(const _Float16* __restrict__ A,
                                 const _Float16* __restrict__ Wh,  // (CC,CC) f16
                                 const float* __restrict__ bias,
                                 const float* __restrict__ bng,
                                 const float* __restrict__ bnb,
                                 const float* __restrict__ bnm,
                                 const float* __restrict__ bnv,
                                 const float* __restrict__ residx,
                                 _Float16* __restrict__ Out) {
  const int n    = blockIdx.y;
  const int m0   = blockIdx.x * 16;
  const int wv   = threadIdx.x >> 5;
  const int lane = threadIdx.x & 31;
  const int l16  = lane & 15;
  const int lh   = lane >> 4;
  const int o    = wv * 16 + l16;     // output channel (B column) for this lane
  const int row  = m0 + l16;          // A row loaded by this lane

  const _Float16* Ab = A + ((size_t)n * TVV + row) * CC;
  const _Float16* Wb = Wh + (size_t)o * CC;
  v8f acc = {};
#pragma unroll
  for (int k0 = 0; k0 < CC; k0 += 32) {
    __builtin_prefetch(Ab + k0 + 32, 0, 1);
    v16h af = *(const v16h*)(Ab + k0 + lh * 16);
    v16h bf = *(const v16h*)(Wb + k0 + lh * 16);
    acc = wmma16(af, bf, acc);
  }
  float scale = 1.0f, shift = bias[o];
  if constexpr (EP) {
    float inv = rsqrtf(bnv[o] + EPS);
    scale = inv * bng[o];
    shift = (bias[o] - bnm[o]) * scale + bnb[o];
  }
#pragma unroll
  for (int r = 0; r < 8; ++r) {
    const int rr = m0 + r + lh * 8;
    const size_t off = ((size_t)n * TVV + rr) * CC + o;
    float val = acc[r] * scale + shift;
    if constexpr (EP) {
      val += residx[off];
      val = val > 0.0f ? val : 0.1f * val;
    }
    Out[off] = (_Float16)val;
  }
}

// ---------------- K2: attention scores (per n,s): Q^T K / (IC*T) -------------
// scores[n,s,u,v] = sum_{t,ic} q[(t+w-1)*V+v', s*16+ic] * k[t*V+v, s*16+ic]
// Edge K-steps peeled; steady-state loop is branch-free with strength-reduced
// incrementing pointers (stride = 2*V*C halves per step).
__global__ void scores_kernel(const _Float16* __restrict__ Kp,
                              const _Float16* __restrict__ Qp,
                              float* __restrict__ Sc) {
  const int s = blockIdx.x, n = blockIdx.y;
  const int wv = threadIdx.x >> 5;           // 0..4  (5 M-tiles of u)
  const int lane = threadIdx.x & 31, l16 = lane & 15, lh = lane >> 4;
  const int u0 = wv * 16;
  const int urow  = u0 + l16;
  const int urowc = urow < UU ? urow : (UU - 1);   // clamped (discarded later)
  const int wwin  = urowc / VV;
  const int vp    = urowc % VV;
  const int c0 = l16;                               // < 25
  const int c1 = (16 + l16) < VV ? (16 + l16) : (VV - 1);  // clamped

  const _Float16* Qb = Qp + (size_t)n * TVV * CC + s * ICC;
  const _Float16* Kb = Kp + (size_t)n * TVV * CC + s * ICC;

  v8f acc0 = {}, acc1 = {};
  const v8h z8 = {};

  auto ldq = [&](int tq) -> v16h {
    return *(const v16h*)(Qb + ((size_t)(tq * VV + vp)) * CC);
  };
  auto ldk = [&](int trow, int c) -> v16h {
    return *(const v16h*)(Kb + ((size_t)(trow * VV + c)) * CC);
  };

  {  // kk = 0 : tq0 = wwin-1 may be -1 (only when wwin==0)
    v16h af = (wwin + lh > 0) ? ldq(wwin - 1 + lh) : mk16(z8, z8);
    v16h b0 = ldk(lh, c0);
    v16h b1 = ldk(lh, c1);
    acc0 = wmma16(af, b0, acc0);
    acc1 = wmma16(af, b1, acc1);
  }
  // steady state: kk in [1, TT/2-2], all addresses advance by a fixed stride
  {
    const size_t stepH = (size_t)2 * VV * CC;  // halves per kk step
    const _Float16* q0p = Qb + ((size_t)((1 + wwin + lh) * VV + vp)) * CC;
    const _Float16* k0p = Kb + ((size_t)((2 + lh) * VV + c0)) * CC;
    const _Float16* k1p = Kb + ((size_t)((2 + lh) * VV + c1)) * CC;
#pragma unroll 2
    for (int kk = 1; kk < TT / 2 - 1; ++kk) {
      v16h af = *(const v16h*)(q0p);
      v16h b0 = *(const v16h*)(k0p);
      v16h b1 = *(const v16h*)(k1p);
      acc0 = wmma16(af, b0, acc0);
      acc1 = wmma16(af, b1, acc1);
      q0p += stepH; k0p += stepH; k1p += stepH;
    }
  }
  {  // kk = TT/2-1 : tq1 = 286+wwin may be 288 (only when wwin==2)
    const int t0 = TT - 2;
    v16h af = (wwin + lh < 3) ? ldq(t0 + wwin - 1 + lh) : mk16(z8, z8);
    v16h b0 = ldk(t0 + lh, c0);
    v16h b1 = ldk(t0 + lh, c1);
    acc0 = wmma16(af, b0, acc0);
    acc1 = wmma16(af, b1, acc1);
  }

  const float scl = 1.0f / (float)(ICC * TT);
  float* out = Sc + ((size_t)(n * SS + s)) * UU * VV;
#pragma unroll
  for (int r = 0; r < 8; ++r) {
    const int u = u0 + r + lh * 8;
    if (u < UU) {
      out[u * VV + l16] = acc0[r] * scl;
      const int v2 = 16 + l16;
      if (v2 < VV) out[u * VV + v2] = acc1[r] * scl;
    }
  }
}

// ---------------- K3: softmax over u, *alpha, +att0; store transposed f16 ---
// AttT layout: (n,s, v[32], u[96]) zero padded
__global__ void softmax_kernel(const float* __restrict__ Sc,
                               const float* __restrict__ alphas,
                               const float* __restrict__ att0,
                               _Float16* __restrict__ AttT) {
  const int s = blockIdx.x, n = blockIdx.y;
  const int v = threadIdx.x;  // 0..31
  _Float16* outrow = AttT + ((size_t)(n * SS + s) * 32 + v) * UPAD;
  if (v < VV) {
    const float* col = Sc + (size_t)(n * SS + s) * UU * VV + v;
    float m = -3.4e38f;
    for (int u = 0; u < UU; ++u) m = fmaxf(m, col[u * VV]);
    float sum = 0.0f;
    for (int u = 0; u < UU; ++u) sum += __expf(col[u * VV] - m);
    const float alpha = alphas[s];
    const float inv = alpha / sum;
    for (int u = 0; u < UPAD; ++u) {
      float val = 0.0f;
      if (u < UU)
        val = __expf(col[u * VV] - m) * inv + att0[(s * UU + u) * VV + v];
      outrow[u] = (_Float16)val;
    }
  } else {
    for (int u = 0; u < UPAD; ++u) outrow[u] = (_Float16)0.0f;
  }
}

// ---------------- K4: fused attention-apply + w_out projection --------------
// per (n,t):  window staged to LDS with async-to-LDS DMA, transposed in LDS;
// for s: y_s = upT(128x96) x att_s(96x32); out += Wout_s x y_s
// epilogue: leaky( x + BN(out + b_out) ) -> y1 (f16)
__global__ void attn_out_kernel(const _Float16* __restrict__ Xh,
                                const _Float16* __restrict__ AttT,
                                const _Float16* __restrict__ Wouth, // (CC,SS*CC) f16
                                const float* __restrict__ bout,
                                const float* __restrict__ g,
                                const float* __restrict__ be,
                                const float* __restrict__ mbn,
                                const float* __restrict__ vbn,
                                const float* __restrict__ Xf,
                                _Float16* __restrict__ Y1) {
  __shared__ _Float16 xwin[UU * CC];   // 18.75 KB, (u, c) row-major staging
  __shared__ _Float16 upT[CC * UPAD];  // 24 KB, (c, u) transposed window
  __shared__ _Float16 yt[32 * CC];     // 8 KB,  (v, c) staging of y_s
  const int t = blockIdx.x, n = blockIdx.y;
  const int tid = threadIdx.x;

  // ---- stage window into LDS via async global->LDS DMA (16B chunks) ----
  {
    const unsigned ldsbase = (unsigned)(size_t)(&xwin[0]);
    const v8h z8 = {};
    for (int q = tid; q < UU * 16; q += 256) {     // 1200 x 16B
      const int u = q >> 4, cc = q & 15;
      const int ts = t + u / VV - 1;
      if (ts >= 0 && ts < TT) {
        const _Float16* gp =
            Xh + ((size_t)n * TVV + ts * VV + (u % VV)) * CC + cc * 8;
        const unsigned lo = ldsbase + (unsigned)(u * CC + cc * 8) * 2;
        asm volatile("global_load_async_to_lds_b128 %0, %1, off"
                     :: "v"(lo), "v"(gp) : "memory");
      } else {
        *(v8h*)(&xwin[u * CC + cc * 8]) = z8;      // zero padding rows
      }
    }
    asm volatile("s_wait_asynccnt 0x0" ::: "memory");
  }
  __syncthreads();
  // ---- build transposed tile (single race-free pass, includes u padding) ---
  for (int e = tid; e < CC * UPAD; e += 256) {
    const int c = e / UPAD, u = e % UPAD;
    upT[e] = (u < UU) ? xwin[u * CC + c] : (_Float16)0.0f;
  }
  __syncthreads();

  const int wv = tid >> 5, lane = tid & 31, l16 = lane & 15, lh = lane >> 4;
  const int m0 = wv * 16;
  v8f acc2a = {}, acc2b = {};

  for (int s = 0; s < SS; ++s) {
    // ---- GEMM 1: y_s = upT x att_s ----
    v8f ya = {}, yb = {};
    const _Float16* attb = AttT + (size_t)((n * SS + s) * 32) * UPAD;
    const int crow = m0 + l16;
#pragma unroll
    for (int kk = 0; kk < 3; ++kk) {
      const int u0k = kk * 32;
      v16h af = *(const v16h*)(upT + crow * UPAD + u0k + lh * 16);
      v16h b0 = *(const v16h*)(attb + (size_t)l16 * UPAD + u0k + lh * 16);
      v16h b1 = *(const v16h*)(attb + (size_t)(16 + l16) * UPAD + u0k + lh * 16);
      ya = wmma16(af, b0, ya);
      yb = wmma16(af, b1, yb);
    }
    __syncthreads();  // protect previous iteration's yt readers
    {
      v8h pk;
#pragma unroll
      for (int r = 0; r < 8; ++r) pk[r] = (_Float16)ya[r];
      *(v8h*)(yt + (size_t)l16 * CC + m0 + lh * 8) = pk;
#pragma unroll
      for (int r = 0; r < 8; ++r) pk[r] = (_Float16)yb[r];
      *(v8h*)(yt + (size_t)(16 + l16) * CC + m0 + lh * 8) = pk;
    }
    __syncthreads();
    // ---- GEMM 2: acc2 += Wout[:, s*128 + c] x yt ----
    const _Float16* wb = Wouth + (size_t)(m0 + l16) * (SS * CC) + s * CC;
#pragma unroll
    for (int k2 = 0; k2 < 4; ++k2) {
      const int c0 = k2 * 32;
      v16h a2 = *(const v16h*)(wb + c0 + lh * 16);
      v16h b2a = *(const v16h*)(yt + (size_t)l16 * CC + c0 + lh * 16);
      v16h b2b = *(const v16h*)(yt + (size_t)(16 + l16) * CC + c0 + lh * 16);
      acc2a = wmma16(a2, b2a, acc2a);
      acc2b = wmma16(a2, b2b, acc2b);
    }
  }
  // ---- epilogue: rows o = m0 + lh*8 + r (contiguous), col v fixed per lane
  const int obase = m0 + lh * 8;
  for (int sel = 0; sel < 2; ++sel) {
    const v8f acc = sel ? acc2b : acc2a;
    const int v = sel * 16 + l16;
    if (v >= VV) continue;
    const size_t rowoff = ((size_t)n * TVV + t * VV + v) * CC;
    v8h pk;
#pragma unroll
    for (int r = 0; r < 8; ++r) {
      const int o = obase + r;
      const float sc = rsqrtf(vbn[o] + EPS) * g[o];
      float val = (acc[r] + bout[o] - mbn[o]) * sc + be[o] + Xf[rowoff + o];
      val = val > 0.0f ? val : 0.1f * val;
      pk[r] = (_Float16)val;
    }
    *(v8h*)(Y1 + rowoff + obase) = pk;
  }
}

// ---------------- K7: pack temporal weights (o, dt, c) f16 ------------------
__global__ void pack_wt_kernel(const float* __restrict__ wt,
                               _Float16* __restrict__ wtp) {
  const int i = blockIdx.x * 256 + threadIdx.x;
  if (i < CC * 7 * CC) {
    const int o = i / (7 * CC);
    const int rem = i % (7 * CC);
    const int dt = rem / CC, c = rem % CC;
    wtp[i] = (_Float16)wt[(size_t)(o * CC + c) * 7 + dt];
  }
}

// ---------------- K6: temporal conv + final epilogue ------------------------
// z[n,tv,o] = sum_{dt,c} y2[n, tv+(dt-3)*V, c] * wtp[o,dt,c]
// out = relu( leaky(y2 + BN(z + b_t)) + x )   (f32)
// Interior blocks (t in [3, T-3)) take a fully unrolled branch-free path with
// all A-loads as immediate offsets from one base pointer.
__global__ void tconv_kernel(const _Float16* __restrict__ Y2,
                             const _Float16* __restrict__ Wtp,
                             const float* __restrict__ bt,
                             const float* __restrict__ g,
                             const float* __restrict__ be,
                             const float* __restrict__ mbn,
                             const float* __restrict__ vbn,
                             const float* __restrict__ Xf,
                             float* __restrict__ Out) {
  const int n = blockIdx.y;
  const int m0 = blockIdx.x * 16;
  const int wv = threadIdx.x >> 5, lane = threadIdx.x & 31;
  const int l16 = lane & 15, lh = lane >> 4;
  const int o = wv * 16 + l16;
  const int row = m0 + l16;

  const _Float16* wrow = Wtp + (size_t)o * (7 * CC);
  v8f acc = {};
  const v8h z8 = {};
  const _Float16* ab0 = Y2 + ((size_t)n * TVV + row - 3 * VV) * CC;

  if (m0 >= 3 * VV && (m0 + 16) <= (TVV - 3 * VV)) {
    // ---- interior: branch-free, immediate-offset loads ----
#pragma unroll
    for (int dt = 0; dt < 7; ++dt) {
#pragma unroll
      for (int kc = 0; kc < 4; ++kc) {
        const int k0 = kc * 32;
        v16h af = *(const v16h*)(ab0 + (size_t)dt * VV * CC + k0 + lh * 16);
        v16h bf = *(const v16h*)(wrow + dt * CC + k0 + lh * 16);
        acc = wmma16(af, bf, acc);
      }
    }
  } else {
    // ---- edge: per-tap mask (zero padding outside [0, T)) ----
    for (int dt = 0; dt < 7; ++dt) {
      const int srow = row + (dt - 3) * VV;
      const bool valid = (srow >= 0) && (srow < TVV);
#pragma unroll
      for (int kc = 0; kc < 4; ++kc) {
        const int k0 = kc * 32;
        v16h af = valid ? *(const v16h*)(ab0 + (size_t)dt * VV * CC + k0 + lh * 16) : mk16(z8, z8);
        v16h bf = *(const v16h*)(wrow + dt * CC + k0 + lh * 16);
        acc = wmma16(af, bf, acc);
      }
    }
  }
  const float scale = rsqrtf(vbn[o] + EPS) * g[o];
  const float shift = (bt[o] - mbn[o]) * scale + be[o];
#pragma unroll
  for (int r = 0; r < 8; ++r) {
    const int rr = m0 + r + lh * 8;
    const size_t off = ((size_t)n * TVV + rr) * CC + o;
    const float y2v = (float)Y2[off];
    float z2 = y2v + acc[r] * scale + shift;
    z2 = z2 > 0.0f ? z2 : 0.1f * z2;
    float outv = z2 + Xf[off];
    Out[off] = outv > 0.0f ? outv : 0.0f;
  }
}

// ---------------- host side ----------------
static inline size_t alignup256(size_t x) { return (x + 255) & ~(size_t)255; }

extern "C" void kernel_launch(void* const* d_in, const int* in_sizes, int n_in,
                              void* d_out, int out_size, void* d_ws,
                              size_t ws_size, hipStream_t stream) {
  const float* x       = (const float*)d_in[0];
  const float* w_k     = (const float*)d_in[1];
  const float* b_k     = (const float*)d_in[2];
  const float* w_q     = (const float*)d_in[3];
  const float* b_q     = (const float*)d_in[4];
  const float* alphas  = (const float*)d_in[5];
  const float* att0    = (const float*)d_in[6];
  const float* w_out   = (const float*)d_in[7];
  const float* b_out   = (const float*)d_in[8];
  const float* g_out   = (const float*)d_in[9];
  const float* be_out  = (const float*)d_in[10];
  const float* m_out   = (const float*)d_in[11];
  const float* v_out   = (const float*)d_in[12];
  const float* w_ff    = (const float*)d_in[13];
  const float* b_ff    = (const float*)d_in[14];
  const float* g_ff    = (const float*)d_in[15];
  const float* be_ff   = (const float*)d_in[16];
  const float* m_ff    = (const float*)d_in[17];
  const float* v_ff    = (const float*)d_in[18];
  const float* w_t     = (const float*)d_in[19];
  const float* b_t     = (const float*)d_in[20];
  const float* g_t     = (const float*)d_in[21];
  const float* be_t    = (const float*)d_in[22];
  const float* m_t     = (const float*)d_in[23];
  const float* v_t     = (const float*)d_in[24];
  float* out = (float*)d_out;

  // workspace layout (y1 aliases kproj, y2 aliases qproj: dead after scores)
  char* ws = (char*)d_ws;
  const size_t actH = (size_t)NB * TVV * CC * sizeof(_Float16);  // 29.5 MB
  size_t off = 0;
  _Float16* xh = (_Float16*)(ws + off); off = alignup256(off + actH);
  _Float16* kp = (_Float16*)(ws + off); off = alignup256(off + actH);
  _Float16* qp = (_Float16*)(ws + off); off = alignup256(off + actH);
  float* sc = (float*)(ws + off);
  off = alignup256(off + (size_t)NB * SS * UU * VV * sizeof(float));
  _Float16* attT = (_Float16*)(ws + off);
  off = alignup256(off + (size_t)NB * SS * 32 * UPAD * sizeof(_Float16));
  _Float16* wtp = (_Float16*)(ws + off);
  off = alignup256(off + (size_t)CC * 7 * CC * sizeof(_Float16));
  _Float16* wkh = (_Float16*)(ws + off);
  off = alignup256(off + (size_t)CC * CC * sizeof(_Float16));
  _Float16* wqh = (_Float16*)(ws + off);
  off = alignup256(off + (size_t)CC * CC * sizeof(_Float16));
  _Float16* wfh = (_Float16*)(ws + off);
  off = alignup256(off + (size_t)CC * CC * sizeof(_Float16));
  _Float16* wouth = (_Float16*)(ws + off);
  off = alignup256(off + (size_t)CC * SS * CC * sizeof(_Float16));
  _Float16* y1 = kp;  // alias
  _Float16* y2 = qp;  // alias
  (void)ws_size; (void)in_sizes; (void)n_in; (void)out_size;

  // ---- packing passes (f16 everything that feeds WMMA) ----
  const int total = NB * TVV * CC;
  f32_to_f16_kernel<<<total / (256 * 4), 256, 0, stream>>>(x, xh, total / 4);
  f32_to_f16_kernel<<<16, 256, 0, stream>>>(w_k, wkh, CC * CC / 4);
  f32_to_f16_kernel<<<16, 256, 0, stream>>>(w_q, wqh, CC * CC / 4);
  f32_to_f16_kernel<<<16, 256, 0, stream>>>(w_ff, wfh, CC * CC / 4);
  f32_to_f16_kernel<<<128, 256, 0, stream>>>(w_out, wouth, CC * SS * CC / 4);
  pack_wt_kernel<<<(CC * 7 * CC + 255) / 256, 256, 0, stream>>>(w_t, wtp);

  // ---- main pipeline ----
  dim3 gg(TVV / 16, NB);
  rowgemm16_kernel<false><<<gg, 256, 0, stream>>>(
      xh, wkh, b_k, nullptr, nullptr, nullptr, nullptr, nullptr, kp);
  rowgemm16_kernel<false><<<gg, 256, 0, stream>>>(
      xh, wqh, b_q, nullptr, nullptr, nullptr, nullptr, nullptr, qp);

  scores_kernel<<<dim3(SS, NB), 160, 0, stream>>>(kp, qp, sc);
  softmax_kernel<<<dim3(SS, NB), 32, 0, stream>>>(sc, alphas, att0, attT);

  attn_out_kernel<<<dim3(TT, NB), 256, 0, stream>>>(
      xh, attT, wouth, b_out, g_out, be_out, m_out, v_out, x, y1);

  rowgemm16_kernel<true><<<gg, 256, 0, stream>>>(
      y1, wfh, b_ff, g_ff, be_ff, m_ff, v_ff, x, y2);

  tconv_kernel<<<gg, 256, 0, stream>>>(
      y2, wtp, b_t, g_t, be_t, m_t, v_t, x, out);
}
